// MHA_9603546874182
// MI455X (gfx1250) — matrix operand order repaired and checked by reading, not compiled, not run on hardware
//
#include <hip/hip_runtime.h>

typedef __bf16 bf16;
typedef __attribute__((ext_vector_type(16))) __bf16 v16bf;
typedef __attribute__((ext_vector_type(8)))  __bf16 v8bf;
typedef __attribute__((ext_vector_type(4)))  __bf16 v4bf;
typedef __attribute__((ext_vector_type(8)))  float  v8f;

#define WMMA_BF16(a, b, c) \
  __builtin_amdgcn_wmma_f32_16x16x32_bf16(false, (a), false, (b), (short)0, (c), false, false)

static __device__ __forceinline__ v16bf cat8(v8bf lo, v8bf hi) {
  return __builtin_shufflevector(lo, hi, 0,1,2,3,4,5,6,7,8,9,10,11,12,13,14,15);
}

// Async DMA: 16B per lane, global -> LDS, tracked by ASYNCcnt (CDNA5).
// lds_off is the wave-relative LDS byte address (low 32 bits of flat shared ptr).
static __device__ __forceinline__ void async_b128(unsigned lds_off, const void* g) {
  asm volatile("global_load_async_to_lds_b128 %0, %1, off"
               :: "v"(lds_off), "v"((unsigned long long)(uintptr_t)g)
               : "memory");
}

// ---------------------------------------------------------------------------
// fp32 -> bf16 conversion, 4 elements / thread
// ---------------------------------------------------------------------------
__global__ void f32_to_bf16_kernel(const float* __restrict__ src,
                                   bf16* __restrict__ dst, int n4) {
  int i = blockIdx.x * blockDim.x + threadIdx.x;
  if (i < n4) {
    float4 f = ((const float4*)src)[i];
    v4bf o;
    o[0] = (bf16)f.x; o[1] = (bf16)f.y; o[2] = (bf16)f.z; o[3] = (bf16)f.w;
    ((v4bf*)dst)[i] = o;
  }
}

// ---------------------------------------------------------------------------
// NT GEMM: C[m][n] = sum_k A[m][k] * B[n][k] (+ bias[n])
// mode 0: write fp32 C (row stride N)
// mode 1: QKV epilogue: scatter bf16 into q/k/v [b][h][t][d]; q scaled 1/8
// Block tile 128x128, BK=32, 8 waves of 32x64.
// Double-buffered LDS fed by async global->LDS DMA.
// ---------------------------------------------------------------------------
__global__ __launch_bounds__(256) void gemm_bf16_nt(
    const bf16* __restrict__ A, const bf16* __restrict__ Bw,
    const float* __restrict__ bias, float* __restrict__ Cf,
    bf16* __restrict__ Qo, bf16* __restrict__ Ko, bf16* __restrict__ Vo,
    int M, int N, int K, int mode) {
  __shared__ __align__(32) bf16 As[2][128 * 32];
  __shared__ __align__(32) bf16 Bs[2][128 * 32];
  const int tid = threadIdx.x;
  const int w = tid >> 5, lane = tid & 31, hf = lane >> 4, l15 = lane & 15;
  const int bm = blockIdx.y * 128, bn = blockIdx.x * 128;
  const int wm = (w & 3) * 32, wn = (w >> 2) * 64;
  v8f acc[2][4] = {};

  const int r = tid >> 1, cb = (tid & 1) * 16;
  const unsigned ldsA0 = (unsigned)(uintptr_t)&As[0][0];
  const unsigned ldsB0 = (unsigned)(uintptr_t)&Bs[0][0];
  const unsigned eoff  = (unsigned)((r * 32 + cb) * 2);

  auto stage = [&](int buf, int k0) {
    const bf16* ga = A  + (size_t)(bm + r) * K + k0 + cb;
    const bf16* gb = Bw + (size_t)(bn + r) * K + k0 + cb;
    const unsigned la = ldsA0 + (unsigned)buf * (128 * 32 * 2) + eoff;
    const unsigned lb = ldsB0 + (unsigned)buf * (128 * 32 * 2) + eoff;
    async_b128(la,      ga);
    async_b128(la + 16, ga + 8);
    async_b128(lb,      gb);
    async_b128(lb + 16, gb + 8);
  };

  const int niter = K / 32;
  stage(0, 0);
  for (int it = 0; it < niter; ++it) {
    if (it + 1 < niter) {
      stage((it + 1) & 1, (it + 1) * 32);   // overlap next tile's DMA
      asm volatile("s_wait_asynccnt 4" ::: "memory");  // tile `it` landed
    } else {
      asm volatile("s_wait_asynccnt 0" ::: "memory");
    }
    __syncthreads();

    const bf16* Ab = As[it & 1];
    const bf16* Bb = Bs[it & 1];
    v16bf af[2], bfr[4];
#pragma unroll
    for (int i = 0; i < 2; ++i) {
      const bf16* ar = &Ab[(wm + i * 16 + l15) * 32];
      af[i] = *(const v16bf*)(ar + hf * 16);
    }
#pragma unroll
    for (int j = 0; j < 4; ++j)
      bfr[j] = *(const v16bf*)&Bb[(wn + j * 16 + l15) * 32 + hf * 16];
#pragma unroll
    for (int i = 0; i < 2; ++i)
#pragma unroll
      for (int j = 0; j < 4; ++j)
        acc[i][j] = WMMA_BF16(af[i], bfr[j], acc[i][j]);

    __syncthreads();  // all reads of this buffer done before it is re-staged
  }

#pragma unroll
  for (int i = 0; i < 2; ++i)
#pragma unroll
    for (int j = 0; j < 4; ++j)
#pragma unroll
      for (int v = 0; v < 8; ++v) {
        const int m = bm + wm + i * 16 + v + 8 * hf;
        const int n = bn + wn + j * 16 + l15;
        float val = acc[i][j][v] + (bias ? bias[n] : 0.0f);
        if (mode == 0) {
          Cf[(size_t)m * N + n] = val;
        } else {
          const int which = n >> 11, c = n & 2047, hh = c >> 6, d = c & 63;
          const int bi = m >> 10, t = m & 1023;
          if (which == 0) val *= 0.125f;  // fold 1/sqrt(hd) into Q
          bf16* dst = (which == 0) ? Qo : ((which == 1) ? Ko : Vo);
          dst[(((size_t)bi * 32 + hh) * 1024 + t) * 64 + d] = (bf16)val;
        }
      }
}

// ---------------------------------------------------------------------------
// RoPE in-place on bf16 Q and K, layout [b][h][t][64], ROT_DIM=32
// ---------------------------------------------------------------------------
__global__ void rope_kernel(bf16* __restrict__ Qm, bf16* __restrict__ Km,
                            const float* __restrict__ rope) {
  int idx = blockIdx.x * blockDim.x + threadIdx.x;
  int p   = idx & 15;
  int t   = (idx >> 4) & 1023;
  int bh  = (idx >> 14) & 127;
  int sel = idx >> 21;
  bf16* ptr = sel ? Km : Qm;
  float cs = rope[t * 32 + p * 2 + 0];
  float sn = rope[t * 32 + p * 2 + 1];
  size_t base = ((size_t)bh * 1024 + t) * 64;
  float x1 = (float)ptr[base + p];
  float x2 = (float)ptr[base + p + 16];
  ptr[base + p]      = (bf16)(x1 * cs - x2 * sn);
  ptr[base + p + 16] = (bf16)(x2 * cs + x1 * sn);
}

// ---------------------------------------------------------------------------
// Flash attention: block = 128 queries for one (b,h); 8 waves x 16 q-rows.
// Online softmax; scores and P.V through WMMA bf16. Q pre-scaled by 1/sqrt(hd).
// Double-buffered K/V tiles fed by async global->LDS DMA.
// ---------------------------------------------------------------------------
__global__ __launch_bounds__(256) void attn_kernel(
    const bf16* __restrict__ Qm, const bf16* __restrict__ Km,
    const bf16* __restrict__ Vm, bf16* __restrict__ ctx) {
  __shared__ __align__(32) bf16 Ks[2][32 * 64];
  __shared__ __align__(32) bf16 Vs[2][32 * 64];
  __shared__ __align__(32) bf16 Ps[8 * 16 * 32];
  const int tid = threadIdx.x;
  const int w = tid >> 5, lane = tid & 31, hf = lane >> 4, l15 = lane & 15;
  const int bh = blockIdx.y;
  const int b = bh >> 5, h = bh & 31;
  const int q0 = blockIdx.x * 128 + w * 16;
  const size_t bhBase = (size_t)bh * 1024 * 64;

  const unsigned ldsK0 = (unsigned)(uintptr_t)&Ks[0][0];
  const unsigned ldsV0 = (unsigned)(uintptr_t)&Vs[0][0];
  const int rr = tid >> 3, ch = (tid & 7) * 8;
  const unsigned eoff = (unsigned)((rr * 64 + ch) * 2);

  auto stageKV = [&](int buf, int j0) {
    const bf16* kg = Km + bhBase + (size_t)(j0 + rr) * 64 + ch;
    const bf16* vg = Vm + bhBase + (size_t)(j0 + rr) * 64 + ch;
    async_b128(ldsK0 + (unsigned)buf * (32 * 64 * 2) + eoff, kg);
    async_b128(ldsV0 + (unsigned)buf * (32 * 64 * 2) + eoff, vg);
  };

  // Q fragments: A layout, lane holds row m=l15, k-chunks per half
  v16bf aq[2];
  {
    const bf16* qrow = Qm + bhBase + (size_t)(q0 + l15) * 64;
#pragma unroll
    for (int f = 0; f < 2; ++f)
      aq[f] = *(const v16bf*)(qrow + f * 32 + hf * 16);
  }

  v8f accO[4] = {};
  float rm[8], rl[8];
#pragma unroll
  for (int v = 0; v < 8; ++v) { rm[v] = -1e30f; rl[v] = 0.0f; }

  const int nkb = (blockIdx.x + 1) * 4;  // uniform causal trip count
  stageKV(0, 0);
  for (int kb = 0; kb < nkb; ++kb) {
    const int j0 = kb * 32;
    if (kb + 1 < nkb) {
      stageKV((kb + 1) & 1, (kb + 1) * 32);
      asm volatile("s_wait_asynccnt 2" ::: "memory");
    } else {
      asm volatile("s_wait_asynccnt 0" ::: "memory");
    }
    __syncthreads();
    const bf16* Kb = Ks[kb & 1];
    const bf16* Vb = Vs[kb & 1];

    // scores: 16q x 32k via 4 WMMAs
    v8f sacc[2] = {};
#pragma unroll
    for (int t = 0; t < 2; ++t)
#pragma unroll
      for (int f = 0; f < 2; ++f) {
        v16bf bk = *(const v16bf*)&Kb[(t * 16 + l15) * 64 + f * 32 + hf * 16];
        sacc[t] = WMMA_BF16(aq[f], bk, sacc[t]);
      }

    // mask + online softmax (row m = v + 8*hf, col n = l15)
#pragma unroll
    for (int v = 0; v < 8; ++v) {
      const int qrow = q0 + v + 8 * hf;
      float s0 = sacc[0][v]; if (j0 + l15 > qrow)      s0 = -1e30f;
      float s1 = sacc[1][v]; if (j0 + 16 + l15 > qrow) s1 = -1e30f;
      float mx = fmaxf(s0, s1);
      mx = fmaxf(mx, __shfl_xor(mx, 1));
      mx = fmaxf(mx, __shfl_xor(mx, 2));
      mx = fmaxf(mx, __shfl_xor(mx, 4));
      mx = fmaxf(mx, __shfl_xor(mx, 8));
      const float rmNew = fmaxf(rm[v], mx);
      const float p0 = __expf(s0 - rmNew);
      const float p1 = __expf(s1 - rmNew);
      const float corr = __expf(rm[v] - rmNew);
      float rs = p0 + p1;
      rs += __shfl_xor(rs, 1);
      rs += __shfl_xor(rs, 2);
      rs += __shfl_xor(rs, 4);
      rs += __shfl_xor(rs, 8);
      rl[v] = rl[v] * corr + rs;
      rm[v] = rmNew;
#pragma unroll
      for (int j = 0; j < 4; ++j) accO[j][v] *= corr;
      bf16* pw = &Ps[(w * 16 + v + 8 * hf) * 32];
      pw[l15]      = (bf16)p0;
      pw[16 + l15] = (bf16)p1;
    }
    __syncthreads();  // C-layout -> A-layout round trip through LDS

    v16bf ap;
    {
      const bf16* pr = &Ps[(w * 16 + l15) * 32];
      ap = *(const v16bf*)(pr + hf * 16);
    }
#pragma unroll
    for (int j = 0; j < 4; ++j) {
      v16bf bv;
#pragma unroll
      for (int i = 0; i < 16; ++i)
        bv[i] = Vb[(hf * 16 + i) * 64 + j * 16 + l15];
      accO[j] = WMMA_BF16(ap, bv, accO[j]);
    }
    __syncthreads();  // done reading Ks/Vs/Ps before re-stage/overwrite
  }

  // normalize + write ctx as bf16 in [b][t][C] layout
  float inv[8];
#pragma unroll
  for (int v = 0; v < 8; ++v) inv[v] = 1.0f / rl[v];
#pragma unroll
  for (int j = 0; j < 4; ++j)
#pragma unroll
    for (int v = 0; v < 8; ++v) {
      const int row = q0 + v + 8 * hf;
      const int col = h * 64 + j * 16 + l15;
      ctx[(size_t)(b * 1024 + row) * 2048 + col] = (bf16)(accO[j][v] * inv[v]);
    }
}

// ---------------------------------------------------------------------------
extern "C" void kernel_launch(void* const* d_in, const int* in_sizes, int n_in,
                              void* d_out, int out_size, void* d_ws,
                              size_t ws_size, hipStream_t stream) {
  (void)in_sizes; (void)n_in; (void)out_size; (void)ws_size;
  const float* x    = (const float*)d_in[0];
  const float* rope = (const float*)d_in[3];
  const float* Wqkv = (const float*)d_in[4];
  const float* bqkv = (const float*)d_in[5];
  const float* Wout = (const float*)d_in[6];
  const float* bout = (const float*)d_in[7];
  float* out = (float*)d_out;

  const int B = 4, T = 1024, C = 2048;
  const int M = B * T;       // 4096
  const int N1 = 3 * C;      // 6144

  char* ws = (char*)d_ws;
  size_t off = 0;
  auto alloc = [&](size_t bytes) {
    char* p = ws + off;
    off += (bytes + 255) & ~(size_t)255;
    return p;
  };
  bf16* xb    = (bf16*)alloc((size_t)M * C * 2);
  bf16* wqkvb = (bf16*)alloc((size_t)N1 * C * 2);
  bf16* woutb = (bf16*)alloc((size_t)C * C * 2);
  bf16* qb    = (bf16*)alloc((size_t)M * C * 2);
  bf16* kb    = (bf16*)alloc((size_t)M * C * 2);
  bf16* vb    = (bf16*)alloc((size_t)M * C * 2);
  bf16* ctxb  = (bf16*)alloc((size_t)M * C * 2);

  int n4 = M * C / 4;
  f32_to_bf16_kernel<<<(n4 + 255) / 256, 256, 0, stream>>>(x, xb, n4);
  n4 = N1 * C / 4;
  f32_to_bf16_kernel<<<(n4 + 255) / 256, 256, 0, stream>>>(Wqkv, wqkvb, n4);
  n4 = C * C / 4;
  f32_to_bf16_kernel<<<(n4 + 255) / 256, 256, 0, stream>>>(Wout, woutb, n4);

  gemm_bf16_nt<<<dim3(N1 / 128, M / 128), 256, 0, stream>>>(
      xb, wqkvb, bqkv, nullptr, qb, kb, vb, M, N1, C, 1);

  rope_kernel<<<(2 * 128 * 1024 * 16) / 256, 256, 0, stream>>>(qb, kb, rope);

  attn_kernel<<<dim3(T / 128, B * 32), 256, 0, stream>>>(qb, kb, vb, ctxb);

  gemm_bf16_nt<<<dim3(C / 128, M / 128), 256, 0, stream>>>(
      ctxb, woutb, bout, out, nullptr, nullptr, nullptr, M, C, C, 0);
}
